// VisionCompressor_84181359001715
// MI455X (gfx1250) — matrix operand order repaired and checked by reading, not compiled, not run on hardware
//
#include <hip/hip_runtime.h>
#include <hip/hip_bf16.h>
#include <stdint.h>

// ---------------------------------------------------------------------------
// Problem constants
// ---------------------------------------------------------------------------
#define BT 8
#define S  576
#define NL 25
#define D  2048
#define CT 1024
#define NLAYER 7
#define HEADS  8
#define HDIM   128
#define ROWS   (BT * S)          // 4608
#define SCALE  0.03125f          // CT^-0.5
#define LN_EPS 1e-5f
#define COS_EPS 1e-8f

// GEMM tiling
#define BM 64
#define BN 128
#define BK 32

typedef __attribute__((ext_vector_type(16))) __bf16 bf16x16;
typedef __attribute__((ext_vector_type(8)))  __bf16 bf16x8;
typedef __attribute__((ext_vector_type(4)))  __bf16 bf16x4;
typedef __attribute__((ext_vector_type(8)))  float  f32x8;

// ---------------------------------------------------------------------------
// CDNA5 async copy: global -> LDS, 16 bytes per lane, tracked by ASYNCcnt.
// ---------------------------------------------------------------------------
__device__ __forceinline__ void async_copy_b128(uint32_t lds_byte_addr,
                                                const void* gaddr) {
    asm volatile("global_load_async_to_lds_b128 %0, %1, off"
                 :: "v"(lds_byte_addr), "v"(gaddr)
                 : "memory");
}
__device__ __forceinline__ void wait_async_le3() {
    asm volatile("s_wait_asynccnt 3" ::: "memory");
}
__device__ __forceinline__ void wait_async_0() {
    asm volatile("s_wait_asynccnt 0" ::: "memory");
}
__device__ __forceinline__ uint32_t lds_addr_of(const void* p) {
    // generic LDS pointer: low 32 bits are the LDS offset (ISA 10.2 aperture)
    return (uint32_t)(uintptr_t)p;
}

// ---------------------------------------------------------------------------
// WMMA fragment loads from LDS tiles (v_wmma_f32_16x16x32_bf16 layout, wave32)
// ---------------------------------------------------------------------------
__device__ __forceinline__ bf16x16 frag_a_lds(const __bf16* row, int kg) {
    bf16x8 lo = *(const bf16x8*)(row + kg * 8);
    bf16x8 hi = *(const bf16x8*)(row + 16 + kg * 8);
    bf16x16 f;
#pragma unroll
    for (int i = 0; i < 8; ++i) { f[i] = lo[i]; f[i + 8] = hi[i]; }
    return f;
}
__device__ __forceinline__ bf16x16 frag_b_lds(const __bf16* col, int kg) {
    bf16x8 lo = *(const bf16x8*)(col + kg * 8);
    bf16x8 hi = *(const bf16x8*)(col + 16 + kg * 8);
    bf16x16 f;
#pragma unroll
    for (int i = 0; i < 8; ++i) { f[i] = lo[i]; f[i + 8] = hi[i]; }
    return f;
}

// ---------------------------------------------------------------------------
// Weight transpose + fp32 -> bf16 convert.  src [K][N] fp32 -> dst [N][K] bf16
// grid = (K/32, N/32), block = (32, 8)
// ---------------------------------------------------------------------------
__global__ void transpose_to_bf16(const float* __restrict__ src,
                                  __bf16* __restrict__ dst, int K, int N) {
    __shared__ float tile[32][33];
    const int k0 = blockIdx.x * 32;
    const int n0 = blockIdx.y * 32;
    const int tx = threadIdx.x;
    const int ty = threadIdx.y;
#pragma unroll
    for (int r = 0; r < 32; r += 8)
        tile[ty + r][tx] = src[(size_t)(k0 + ty + r) * N + (n0 + tx)];
    __syncthreads();
#pragma unroll
    for (int r = 0; r < 32; r += 8)
        dst[(size_t)(n0 + ty + r) * K + (k0 + tx)] = (__bf16)tile[tx][ty + r];
}

// ---------------------------------------------------------------------------
// LayerNorm + bf16 convert.  grid = (ROWS, 8), block = 256.
// ---------------------------------------------------------------------------
__global__ void ln_kernel(const float* __restrict__ feat,
                          const float* __restrict__ ln_scale,
                          const float* __restrict__ ln_bias,
                          __bf16* __restrict__ xnbuf) {
    const int row = blockIdx.x;
    const int l   = blockIdx.y;
    const int layer = (l < NLAYER) ? (24 - 4 * l) : 24;
    const float* x = feat + ((size_t)row * NL + layer) * D;
    __bf16* out = xnbuf + ((size_t)l * ROWS + (size_t)row) * D;
    const int t = threadIdx.x;

    float4 a = *(const float4*)(x + t * 4);
    float4 b = *(const float4*)(x + 1024 + t * 4);

    if (l == NLAYER) {  // raw last layer -> bf16 (for w_in GEMM)
        bf16x4 o0 = {(__bf16)a.x, (__bf16)a.y, (__bf16)a.z, (__bf16)a.w};
        bf16x4 o1 = {(__bf16)b.x, (__bf16)b.y, (__bf16)b.z, (__bf16)b.w};
        *(bf16x4*)(out + t * 4)        = o0;
        *(bf16x4*)(out + 1024 + t * 4) = o1;
        return;
    }

    float s  = a.x + a.y + a.z + a.w + b.x + b.y + b.z + b.w;
    float s2 = a.x*a.x + a.y*a.y + a.z*a.z + a.w*a.w
             + b.x*b.x + b.y*b.y + b.z*b.z + b.w*b.w;

    __shared__ float sh1[256];
    __shared__ float sh2[256];
    sh1[t] = s; sh2[t] = s2;
    __syncthreads();
#pragma unroll
    for (int st = 128; st > 0; st >>= 1) {
        if (t < st) { sh1[t] += sh1[t + st]; sh2[t] += sh2[t + st]; }
        __syncthreads();
    }
    const float mu   = sh1[0] * (1.0f / D);
    const float var  = sh2[0] * (1.0f / D) - mu * mu;
    const float rstd = rsqrtf(var + LN_EPS);

    const float* sc = ln_scale + (size_t)l * D;
    const float* bs = ln_bias  + (size_t)l * D;
    float4 s0 = *(const float4*)(sc + t * 4);
    float4 s1 = *(const float4*)(sc + 1024 + t * 4);
    float4 g0 = *(const float4*)(bs + t * 4);
    float4 g1 = *(const float4*)(bs + 1024 + t * 4);

    bf16x4 o0 = {(__bf16)((a.x - mu) * rstd * s0.x + g0.x),
                 (__bf16)((a.y - mu) * rstd * s0.y + g0.y),
                 (__bf16)((a.z - mu) * rstd * s0.z + g0.z),
                 (__bf16)((a.w - mu) * rstd * s0.w + g0.w)};
    bf16x4 o1 = {(__bf16)((b.x - mu) * rstd * s1.x + g1.x),
                 (__bf16)((b.y - mu) * rstd * s1.y + g1.y),
                 (__bf16)((b.z - mu) * rstd * s1.z + g1.z),
                 (__bf16)((b.w - mu) * rstd * s1.w + g1.w)};
    *(bf16x4*)(out + t * 4)        = o0;
    *(bf16x4*)(out + 1024 + t * 4) = o1;
}

// ---------------------------------------------------------------------------
// Shared GEMM core: C[BMxBN] = A[BMxK] * B^T-layout[BNxK], bf16 in, f32 acc.
// Block = 256 threads / 8 waves (4 m-subtiles x 2 n-subtiles); each wave owns
// a 16x64 sub-tile (4 accumulators).  Triple-buffered LDS tiles filled with
// global_load_async_to_lds_b128, 2-stage-ahead prefetch, ONE barrier/stage:
//   stage kt:  s_wait_asynccnt 3  (stage kt's copies landed, kt+1 in flight)
//              barrier            (also proves all waves did compute(kt-1))
//              prefetch stage kt+2 -> buffer (kt+2)%3 == stage kt-1's buffer
//              compute buffer kt%3
// Steady state unrolled x3 so all buffer indices are compile-time.
// ---------------------------------------------------------------------------
struct GemmAcc { f32x8 acc[4]; };

template <int K>
__device__ __forceinline__ void gemm_core(const __bf16* __restrict__ A,
                                          const __bf16* __restrict__ Bt,
                                          size_t blockM, size_t blockN,
                                          __bf16* lsA, __bf16* lsB,  // [3][BM*BK], [3][BN*BK]
                                          GemmAcc& o) {
    constexpr int KT = K / BK;                 // 64 or 32
    static_assert(KT >= 4, "need at least 4 k-stages");
    const int t    = threadIdx.x;
    const int wave = t >> 5;
    const int lane = t & 31;
    const int mw   = wave & 3;                 // m sub-tile of 16
    const int nw   = wave >> 2;                // n sub-tile of 64
    const int kg   = lane >> 4;
    const int ml   = lane & 15;

    // staging coordinates: 16B (8 bf16) chunks
    const int arow = t >> 2, apart = (t & 3) * 8;   // A: 64 rows x 4 chunks
    const int brow = t >> 2, bpart = (t & 3) * 8;   // B: chunks t and t+256

    const __bf16* gA  = A  + (blockM + arow) * (size_t)K + apart;
    const __bf16* gB0 = Bt + (blockN + brow) * (size_t)K + bpart;
    const __bf16* gB1 = Bt + (blockN + brow + 64) * (size_t)K + bpart;
    const uint32_t lsAaddr = lds_addr_of(lsA) + (uint32_t)t * 16;
    const uint32_t lsBaddr = lds_addr_of(lsB) + (uint32_t)t * 16;

    auto prefetch = [&](int buf, int k0) {
        async_copy_b128(lsAaddr + buf * (BM * BK * 2), gA + k0);
        async_copy_b128(lsBaddr + buf * (BN * BK * 2), gB0 + k0);
        async_copy_b128(lsBaddr + buf * (BN * BK * 2) + 256 * 16, gB1 + k0);
    };

    const __bf16* laBase = lsA + (mw * 16 + ml) * BK;
    const __bf16* lbBase = lsB + (nw * 64 + ml) * BK;

    auto compute = [&](int buf) {
        const __bf16* la = laBase + buf * (BM * BK);
        const __bf16* lb = lbBase + buf * (BN * BK);
        bf16x16 af = frag_a_lds(la, kg);
#pragma unroll
        for (int j = 0; j < 4; ++j) {
            bf16x16 bfr = frag_b_lds(lb + j * 16 * BK, kg);
            o.acc[j] = __builtin_amdgcn_wmma_f32_16x16x32_bf16(
                false, af, false, bfr, (short)0, o.acc[j], false, false);
        }
    };

    prefetch(0, 0);
    prefetch(1, BK);
    int kpre = 2 * BK;

    constexpr int FULL = KT - 2;     // stages that also prefetch
    constexpr int G    = FULL / 3;
    constexpr int R    = FULL % 3;   // 0 (K=1024) or 2 (K=2048)

    for (int g = 0; g < G; ++g) {
        wait_async_le3(); __syncthreads();
        prefetch(2, kpre); kpre += BK;
        compute(0);
        wait_async_le3(); __syncthreads();
        prefetch(0, kpre); kpre += BK;
        compute(1);
        wait_async_le3(); __syncthreads();
        prefetch(1, kpre); kpre += BK;
        compute(2);
    }
    if constexpr (R >= 1) {
        wait_async_le3(); __syncthreads();
        prefetch(2, kpre); kpre += BK;
        compute(0);
    }
    if constexpr (R >= 2) {
        wait_async_le3(); __syncthreads();
        prefetch(0, kpre); kpre += BK;
        compute(1);
    }
    // drain stages KT-2 (buffer R) and KT-1 (buffer (R+1)%3)
    wait_async_le3(); __syncthreads();
    compute(R);
    wait_async_0();  __syncthreads();
    compute((R + 1) % 3);
}

// ---------------------------------------------------------------------------
// Batched GEMM (8 gemms, M=4608 K=2048 N=1024).
// g<7 : v[row][g][:] = xn_g @ w_norm_g      g==7: vin = feat_last @ w_in + b_in
// grid = (72, 8, 8), block = 256.
// ---------------------------------------------------------------------------
__global__ void __launch_bounds__(256, 1)
gemm_v_kernel(const __bf16* __restrict__ xnbuf,
              const __bf16* __restrict__ wT,
              const float* __restrict__ b_in,
              float* __restrict__ vbuf,
              float* __restrict__ vinbuf) {
    __shared__ __bf16 lsA[3][BM * BK];   // 3 x 4 KB
    __shared__ __bf16 lsB[3][BN * BK];   // 3 x 8 KB
    const int g = blockIdx.z;
    const size_t blockM = (size_t)blockIdx.x * BM;
    const size_t blockN = (size_t)blockIdx.y * BN;

    GemmAcc o = {};
    gemm_core<D>(xnbuf + (size_t)g * ROWS * D, wT + (size_t)g * CT * D,
                 blockM, blockN, &lsA[0][0], &lsB[0][0], o);

    const int wave = threadIdx.x >> 5;
    const int lane = threadIdx.x & 31;
    const int m0 = (int)blockM + (wave & 3) * 16;
    const int n0 = (int)blockN + (wave >> 2) * 64;
    const int kgc = lane >> 4;
    const int nl  = lane & 15;
    if (g < NLAYER) {
#pragma unroll
        for (int j = 0; j < 4; ++j) {
            const int col = n0 + j * 16 + nl;
#pragma unroll
            for (int r = 0; r < 8; ++r) {
                const int row = m0 + r + 8 * kgc;
                vbuf[(size_t)row * (NLAYER * CT) + (size_t)g * CT + col] = o.acc[j][r];
            }
        }
    } else {
#pragma unroll
        for (int j = 0; j < 4; ++j) {
            const int col = n0 + j * 16 + nl;
            const float bc = b_in[col];
#pragma unroll
            for (int r = 0; r < 8; ++r) {
                const int row = m0 + r + 8 * kgc;
                vinbuf[(size_t)row * CT + col] = o.acc[j][r] + bc;
            }
        }
    }
}

// ---------------------------------------------------------------------------
// Attention fuse: wave per (row, head).  grid = ROWS, block = 256.
// ---------------------------------------------------------------------------
__global__ void attn_kernel(const float* __restrict__ vbuf,
                            __bf16* __restrict__ fusebuf) {
    const int row  = blockIdx.x;
    const int h    = threadIdx.x >> 5;
    const int lane = threadIdx.x & 31;
    const float* base = vbuf + (size_t)row * (NLAYER * CT) + h * HDIM + lane * 4;

    float4 q = *(const float4*)base;
    float4 kv[6];
    float logit[6];
#pragma unroll
    for (int l = 0; l < 6; ++l) {
        kv[l] = *(const float4*)(base + (size_t)(l + 1) * CT);
        float p = q.x * kv[l].x + q.y * kv[l].y + q.z * kv[l].z + q.w * kv[l].w;
#pragma unroll
        for (int off = 16; off >= 1; off >>= 1) p += __shfl_xor(p, off, 32);
        logit[l] = p * SCALE;
    }
    float m = logit[0];
#pragma unroll
    for (int l = 1; l < 6; ++l) m = fmaxf(m, logit[l]);
    float e[6], ssum = 0.0f;
#pragma unroll
    for (int l = 0; l < 6; ++l) { e[l] = __expf(logit[l] - m); ssum += e[l]; }
    const float inv = 1.0f / ssum;

    float fx = 0, fy = 0, fz = 0, fw = 0;
#pragma unroll
    for (int l = 0; l < 6; ++l) {
        const float a = e[l] * inv;
        fx += a * kv[l].x; fy += a * kv[l].y; fz += a * kv[l].z; fw += a * kv[l].w;
    }
    bf16x4 ov = {(__bf16)fx, (__bf16)fy, (__bf16)fz, (__bf16)fw};
    *(bf16x4*)(fusebuf + (size_t)row * CT + h * HDIM + lane * 4) = ov;
}

// ---------------------------------------------------------------------------
// vf = vin + fuse @ w_out + b_out.  M=4608 K=1024 N=1024.  grid = (72, 8).
// ---------------------------------------------------------------------------
__global__ void __launch_bounds__(256, 1)
gemm_out_kernel(const __bf16* __restrict__ fusebuf,
                const __bf16* __restrict__ wOutT,
                const float* __restrict__ vinbuf,
                const float* __restrict__ b_out,
                float* __restrict__ vfbuf) {
    __shared__ __bf16 lsA[3][BM * BK];
    __shared__ __bf16 lsB[3][BN * BK];
    const size_t blockM = (size_t)blockIdx.x * BM;
    const size_t blockN = (size_t)blockIdx.y * BN;

    GemmAcc o = {};
    gemm_core<CT>(fusebuf, wOutT, blockM, blockN, &lsA[0][0], &lsB[0][0], o);

    const int wave = threadIdx.x >> 5;
    const int lane = threadIdx.x & 31;
    const int m0 = (int)blockM + (wave & 3) * 16;
    const int n0 = (int)blockN + (wave >> 2) * 64;
    const int kgc = lane >> 4;
    const int nl  = lane & 15;
#pragma unroll
    for (int j = 0; j < 4; ++j) {
        const int col = n0 + j * 16 + nl;
        const float bc = b_out[col];
#pragma unroll
        for (int r = 0; r < 8; ++r) {
            const int row = m0 + r + 8 * kgc;
            vfbuf[(size_t)row * CT + col] =
                o.acc[j][r] + vinbuf[(size_t)row * CT + col] + bc;
        }
    }
}

// ---------------------------------------------------------------------------
// Cosine loss per row + deterministic final reduce.
// ---------------------------------------------------------------------------
__global__ void cosine_kernel(const float* __restrict__ vf,
                              const float* __restrict__ teacher,
                              float* __restrict__ rowloss) {
    const int row = blockIdx.x;
    const int t   = threadIdx.x;
    const float* a = vf      + (size_t)row * CT;
    const float* b = teacher + (size_t)row * CT;
    float4 av = *(const float4*)(a + t * 4);
    float4 bv = *(const float4*)(b + t * 4);
    float dot = av.x*bv.x + av.y*bv.y + av.z*bv.z + av.w*bv.w;
    float na  = av.x*av.x + av.y*av.y + av.z*av.z + av.w*av.w;
    float nb  = bv.x*bv.x + bv.y*bv.y + bv.z*bv.z + bv.w*bv.w;

    __shared__ float sd[256], s1[256], s2[256];
    sd[t] = dot; s1[t] = na; s2[t] = nb;
    __syncthreads();
#pragma unroll
    for (int st = 128; st > 0; st >>= 1) {
        if (t < st) { sd[t] += sd[t+st]; s1[t] += s1[t+st]; s2[t] += s2[t+st]; }
        __syncthreads();
    }
    if (t == 0) {
        const float n1 = fmaxf(sqrtf(s1[0]), COS_EPS);
        const float n2 = fmaxf(sqrtf(s2[0]), COS_EPS);
        rowloss[row] = 1.0f - sd[0] / (n1 * n2);
    }
}

__global__ void finalize_kernel(const float* __restrict__ rowloss,
                                float* __restrict__ out) {
    __shared__ float sh[256];
    const int t = threadIdx.x;
    float s = 0.0f;
    for (int i = t; i < ROWS; i += 256) s += rowloss[i];
    sh[t] = s;
    __syncthreads();
#pragma unroll
    for (int st = 128; st > 0; st >>= 1) {
        if (t < st) sh[t] += sh[t + st];
        __syncthreads();
    }
    if (t == 0) out[0] = sh[0] * (1.0f / (float)ROWS);
}

// ---------------------------------------------------------------------------
// Launch
// ---------------------------------------------------------------------------
extern "C" void kernel_launch(void* const* d_in, const int* in_sizes, int n_in,
                              void* d_out, int out_size, void* d_ws, size_t ws_size,
                              hipStream_t stream) {
    (void)in_sizes; (void)n_in; (void)out_size; (void)ws_size;
    const float* features = (const float*)d_in[0];
    const float* teacher  = (const float*)d_in[1];
    const float* ln_scale = (const float*)d_in[2];
    const float* ln_bias  = (const float*)d_in[3];
    const float* w_norm   = (const float*)d_in[4];
    const float* w_in     = (const float*)d_in[5];
    const float* b_in     = (const float*)d_in[6];
    const float* w_out    = (const float*)d_in[7];
    const float* b_out    = (const float*)d_in[8];

    char* p = (char*)d_ws;
    __bf16* xnbuf  = (__bf16*)p; p += (size_t)8 * ROWS * D * 2;       // 151.0 MB
    __bf16* wT     = (__bf16*)p; p += (size_t)8 * CT * D * 2;         //  33.6 MB
    __bf16* wOutT  = (__bf16*)p; p += (size_t)CT * CT * 2;            //   2.1 MB
    float*  vbuf   = (float*)p;  p += (size_t)ROWS * NLAYER * CT * 4; // 132.1 MB
    float*  vinbuf = (float*)p;  p += (size_t)ROWS * CT * 4;          //  18.9 MB
    __bf16* fusebuf= (__bf16*)p; p += (size_t)ROWS * CT * 2;          //   9.4 MB
    float*  vfbuf  = (float*)p;  p += (size_t)ROWS * CT * 4;          //  18.9 MB
    float*  rowloss= (float*)p;  p += (size_t)ROWS * 4;

    const dim3 tb(32, 8);
    for (int g = 0; g < NLAYER; ++g)
        transpose_to_bf16<<<dim3(D / 32, CT / 32), tb, 0, stream>>>(
            w_norm + (size_t)g * D * CT, wT + (size_t)g * CT * D, D, CT);
    transpose_to_bf16<<<dim3(D / 32, CT / 32), tb, 0, stream>>>(
        w_in, wT + (size_t)NLAYER * CT * D, D, CT);
    transpose_to_bf16<<<dim3(CT / 32, CT / 32), tb, 0, stream>>>(
        w_out, wOutT, CT, CT);

    ln_kernel<<<dim3(ROWS, 8), 256, 0, stream>>>(features, ln_scale, ln_bias, xnbuf);

    gemm_v_kernel<<<dim3(ROWS / BM, CT / BN, 8), 256, 0, stream>>>(
        xnbuf, wT, b_in, vbuf, vinbuf);

    attn_kernel<<<ROWS, 256, 0, stream>>>(vbuf, fusebuf);

    gemm_out_kernel<<<dim3(ROWS / BM, CT / BN), 256, 0, stream>>>(
        fusebuf, wOutT, vinbuf, b_out, vfbuf);

    cosine_kernel<<<ROWS, 256, 0, stream>>>(vfbuf, teacher, rowloss);
    finalize_kernel<<<1, 256, 0, stream>>>(rowloss, (float*)d_out);
}
